// GraphCastNet_24507083391118
// MI455X (gfx1250) — compile-verified, matrix operand order checked by reading, not run on hardware
//
#include <hip/hip_runtime.h>
#include <cstdint>
#include <cstddef>

// ---------------- problem constants ----------------
#define HID      512
#define N_GRID_C 65160
#define N_MESH_C 40962
#define E_G2M_C  260640
#define E_MESH_C 327660
#define E_M2G_C  195480

typedef __bf16 bf16;
typedef __attribute__((ext_vector_type(16))) __bf16 bf16x16;
typedef __attribute__((ext_vector_type(8)))  __bf16 bf16x8;
typedef __attribute__((ext_vector_type(8)))  float  f32x8;

// ---------------- CDNA5 async global->LDS path (guarded) ----------------
#if defined(__HIP_DEVICE_COMPILE__)
#if __has_builtin(__builtin_amdgcn_global_load_async_to_lds_b64)
#define HAVE_ASYNC_LDS 1
#endif
#endif

typedef int v2i __attribute__((vector_size(8)));
typedef __attribute__((address_space(1))) v2i* as1_v2i;   // global v2i*
typedef __attribute__((address_space(3))) v2i* as3_v2i;   // LDS v2i*

static __device__ __forceinline__ void lds_async_cp8(void* lds_dst, const void* gsrc) {
#ifdef HAVE_ASYNC_LDS
  __builtin_amdgcn_global_load_async_to_lds_b64((as1_v2i)gsrc, (as3_v2i)lds_dst, 0, 0);
#else
  *(uint2*)lds_dst = *(const uint2*)gsrc;
#endif
}
static __device__ __forceinline__ void lds_async_wait() {
#ifdef HAVE_ASYNC_LDS
#if __has_builtin(__builtin_amdgcn_s_wait_asynccnt)
  __builtin_amdgcn_s_wait_asynccnt(0);
#else
  asm volatile("s_wait_asynccnt 0" ::: "memory");
#endif
#endif
}

// ---------------- small device helpers ----------------
static __device__ __forceinline__ bf16 f2bf(float f) {
  unsigned u = __float_as_uint(f);
  u += 0x7FFFu + ((u >> 16) & 1u);          // round-to-nearest-even
  unsigned short h = (unsigned short)(u >> 16);
  return __builtin_bit_cast(bf16, h);
}
static __device__ __forceinline__ float bf2f(bf16 b) {
  unsigned short h = __builtin_bit_cast(unsigned short, b);
  return __uint_as_float(((unsigned)h) << 16);
}
// Build a 16-elem bf16 WMMA fragment from two 16B halves (K 0..7 / K 16..23 per lane)
static __device__ __forceinline__ bf16x16 frag_ld(const bf16* p0, const bf16* p1) {
  bf16x8 lo = *(const bf16x8*)p0;
  bf16x8 hi = *(const bf16x8*)p1;
  return __builtin_shufflevector(lo, hi, 0,1,2,3,4,5,6,7,8,9,10,11,12,13,14,15);
}

// ---------------- weight transpose+convert: w[K][N] f32 -> wt[N][Kpad] bf16 ----------------
__global__ void k_wT(const float* __restrict__ w, bf16* __restrict__ wt,
                     int K, int N, int Kpad) {
  int n = blockIdx.x;
  for (int k = threadIdx.x; k < Kpad; k += blockDim.x)
    wt[(size_t)n * Kpad + k] = (k < K) ? f2bf(w[(size_t)k * N + n]) : f2bf(0.f);
}

// ---------------- zero fill ----------------
__global__ void k_zero(float* __restrict__ p, long long n) {
  long long i = (long long)blockIdx.x * blockDim.x + threadIdx.x;
  if (i < n) p[i] = 0.f;
}

// ---------------- segment-sum scatter: agg[dst[e]] += e_feat[e] ----------------
__global__ void k_scatter(const bf16* __restrict__ e, const int* __restrict__ dst,
                          float* __restrict__ agg, long long total) {
  long long g = (long long)blockIdx.x * blockDim.x + threadIdx.x;
  if (g >= total) return;
  int col = (int)(g & (HID - 1));
  long long row = g >> 9;              // HID == 512
  int d = dst[row];
  float v = bf2f(e[g]);
  __hip_atomic_fetch_add(&agg[(size_t)d * HID + col], v,
                         __ATOMIC_RELAXED, __HIP_MEMORY_SCOPE_AGENT);
}

// ---------------- fused gather-concat MLP (din->512 SiLU ->512, LN, residual) ----------------
struct MlpArgs {
  const bf16* s0b; const float* s0f; int d0;     // segment 0: bf16 activations OR f32 raw/agg
  const bf16* s1;  const int* i1;    int d1;     // segment 1: optional gather (idx==null -> row)
  const bf16* s2;  const int* i2;    int d2;     // segment 2: optional gather
  int din, k1pad, nrows;
  const bf16* w0T; const float* b0;              // w0T: [HID][k1pad] bf16
  const bf16* w1T; const float* b1;              // w1T: [HID][HID]  bf16
  const float* lng; const float* lnb;
  const bf16* resid;                             // nullable (may alias out)
  bf16* out;
};

static __device__ __forceinline__ bf16 gather_in(const MlpArgs& a, int row, int col) {
  if (row >= a.nrows || col >= a.din) return f2bf(0.f);
  if (col < a.d0) {
    if (a.s0f) return f2bf(a.s0f[(size_t)row * a.d0 + col]);
    return a.s0b[(size_t)row * a.d0 + col];
  }
  col -= a.d0;
  if (col < a.d1) {
    int r = a.i1 ? a.i1[row] : row;
    return a.s1[(size_t)r * a.d1 + col];
  }
  col -= a.d1;
  int r = a.i2 ? a.i2[row] : row;
  return a.s2[(size_t)r * a.d2 + col];
}

// pointer to 4 consecutive bf16 of the concatenated input (all-bf16 segments only)
static __device__ __forceinline__ const bf16* seg_ptr(const MlpArgs& a, int row, int col) {
  if (col < a.d0) return a.s0b + (size_t)row * a.d0 + col;
  col -= a.d0;
  if (col < a.d1) { int r = a.i1 ? a.i1[row] : row; return a.s1 + (size_t)r * a.d1 + col; }
  col -= a.d1;
  int r = a.i2 ? a.i2[row] : row;
  return a.s2 + (size_t)r * a.d2 + col;
}

__global__ __launch_bounds__(256) void k_mlp(MlpArgs a) {
  __shared__ bf16  sA[32 * 40];     // staged A K-slab, row stride 40 (16B aligned)
  __shared__ bf16  sH[32 * 520];    // hidden / output rows (bf16), stride 520
  __shared__ float sRed[512];       // LN reduction scratch

  const int tid  = threadIdx.x;
  const int lane = tid & 31;
  const int wv   = tid >> 5;        // 8 waves
  const int wm   = wv >> 2;         // 0..1 : M-tile
  const int wn   = wv & 3;          // 0..3 : N-tile (128 cols each)
  const int lm   = lane & 15;
  const int lk   = (lane >> 4) << 3;   // K half select (0 or 8)
  const int rsel = (lane >> 4) << 3;   // C-frag row offset (0 or 8)
  const int row0 = blockIdx.x * 32;

  f32x8 acc[8];
  for (int j = 0; j < 8; ++j) for (int q = 0; q < 8; ++q) acc[j][q] = 0.f;

  // hoisted per-wave fragment base pointers (loop-invariant 64-bit math)
  const bf16* bpj0[8];
  for (int j = 0; j < 8; ++j)
    bpj0[j] = a.w0T + (size_t)(wn * 128 + j * 16 + lm) * a.k1pad + lk;
  const bf16* apA = &sA[(wm * 16 + lm) * 40 + lk];

  // ---------------- GEMM1: y1 = X @ W0 ----------------
  const int nK1 = a.k1pad >> 5;
  const bool all_bf16 = (a.s0f == nullptr);
  for (int ks = 0; ks < nK1; ++ks) {
    __syncthreads();
    const int kb = ks << 5;
    if (all_bf16) {
      // CDNA5 async global->LDS staging: 256 lanes x 8B chunks = 32x32 bf16 tile
      int r    = tid >> 3;
      int c4   = (tid & 7) << 2;      // 0,4,...,28
      int grow = row0 + r;
      bf16* ld = &sA[r * 40 + c4];
      if (grow < a.nrows) {
        lds_async_cp8(ld, seg_ptr(a, grow, kb + c4));
      } else {
        uint2 z; z.x = 0u; z.y = 0u;
        *(uint2*)ld = z;
      }
      lds_async_wait();
    } else {
      for (int i = 0; i < 4; ++i) {               // scalar gather+convert path
        int e = tid + (i << 8);
        int r = e >> 5, kk = e & 31;
        sA[r * 40 + kk] = gather_in(a, row0 + r, kb + kk);
      }
    }
    __syncthreads();
    bf16x16 af = frag_ld(apA, apA + 16);
#pragma unroll
    for (int j = 0; j < 8; ++j) {
      const bf16* bp = bpj0[j] + kb;
      bf16x16 bv = frag_ld(bp, bp + 16);
      acc[j] = __builtin_amdgcn_wmma_f32_16x16x32_bf16(
          false, af, false, bv, (short)0, acc[j], false, false);
    }
  }

  // bias + SiLU -> sH (bf16)
  for (int j = 0; j < 8; ++j) {
    int col = wn * 128 + j * 16 + lm;
    float bb = a.b0[col];
    for (int q = 0; q < 8; ++q) {
      float v = acc[j][q] + bb;
      v = v * (1.f / (1.f + __expf(-v)));
      sH[(wm * 16 + q + rsel) * 520 + col] = f2bf(v);
    }
  }
  __syncthreads();

  // ---------------- GEMM2: y = H @ W1 (K=512, fully unrolled: imm offsets) ----------------
  for (int j = 0; j < 8; ++j) for (int q = 0; q < 8; ++q) acc[j][q] = 0.f;
  {
    const bf16* bpj1[8];
    for (int j = 0; j < 8; ++j)
      bpj1[j] = a.w1T + (size_t)(wn * 128 + j * 16 + lm) * HID + lk;
    const bf16* apH = &sH[(wm * 16 + lm) * 520 + lk];
#pragma unroll
    for (int ks = 0; ks < 16; ++ks) {
      const int kb = ks << 5;
      bf16x16 af = frag_ld(apH + kb, apH + kb + 16);
#pragma unroll
      for (int j = 0; j < 8; ++j) {
        bf16x16 bv = frag_ld(bpj1[j] + kb, bpj1[j] + kb + 16);
        acc[j] = __builtin_amdgcn_wmma_f32_16x16x32_bf16(
            false, af, false, bv, (short)0, acc[j], false, false);
      }
    }
  }
  __syncthreads();                                // all sH reads done; reuse as y buffer
  for (int j = 0; j < 8; ++j) {
    int col = wn * 128 + j * 16 + lm;
    float bb = a.b1[col];
    for (int q = 0; q < 8; ++q)
      sH[(wm * 16 + q + rsel) * 520 + col] = f2bf(acc[j][q] + bb);
  }
  __syncthreads();

  // ---------------- LayerNorm + residual + store ----------------
  int r    = tid >> 3;          // 32 rows, 8 threads/row
  int part = tid & 7;
  float s = 0.f, sq = 0.f;
  for (int c = part * 64; c < part * 64 + 64; ++c) {
    float v = bf2f(sH[r * 520 + c]);
    s += v; sq += v * v;
  }
  sRed[tid] = s; sRed[256 + tid] = sq;
  __syncthreads();
  float ts = 0.f, tq = 0.f;
  for (int i = 0; i < 8; ++i) { ts += sRed[(r << 3) + i]; tq += sRed[256 + (r << 3) + i]; }
  float mu   = ts * (1.f / 512.f);
  float var  = tq * (1.f / 512.f) - mu * mu;
  float rstd = rsqrtf(var + 1e-5f);
  int grow = row0 + r;
  if (grow < a.nrows) {
    for (int c = part * 64; c < part * 64 + 64; ++c) {
      float v = (bf2f(sH[r * 520 + c]) - mu) * rstd * a.lng[c] + a.lnb[c];
      if (a.resid) v += bf2f(a.resid[(size_t)grow * HID + c]);
      a.out[(size_t)grow * HID + c] = f2bf(v);
    }
  }
}

// ---------------- final head: 512 -> 512 (SiLU, WMMA) -> 8 (VALU dot), f32 out ----------------
struct FinArgs {
  const bf16* x;       // [nrows][512] bf16
  const bf16* w0T;     // [512][512] bf16
  const float* b0;
  const float* w1;     // [512][8] f32
  const float* b1;     // [8]
  float* out;          // [nrows][8]
  int nrows;
};

__global__ __launch_bounds__(256) void k_final(FinArgs a) {
  __shared__ bf16 sA[32 * 40];
  __shared__ bf16 sH[32 * 520];

  const int tid  = threadIdx.x;
  const int lane = tid & 31;
  const int wv   = tid >> 5;
  const int wm   = wv >> 2;
  const int wn   = wv & 3;
  const int lm   = lane & 15;
  const int lk   = (lane >> 4) << 3;
  const int rsel = (lane >> 4) << 3;
  const int row0 = blockIdx.x * 32;

  f32x8 acc[8];
  for (int j = 0; j < 8; ++j) for (int q = 0; q < 8; ++q) acc[j][q] = 0.f;

  const bf16* bpj[8];
  for (int j = 0; j < 8; ++j)
    bpj[j] = a.w0T + (size_t)(wn * 128 + j * 16 + lm) * HID + lk;
  const bf16* apA = &sA[(wm * 16 + lm) * 40 + lk];

  for (int ks = 0; ks < 16; ++ks) {
    __syncthreads();
    const int kb = ks << 5;
    {
      int r    = tid >> 3;
      int c4   = (tid & 7) << 2;
      int grow = row0 + r;
      bf16* ld = &sA[r * 40 + c4];
      if (grow < a.nrows) {
        lds_async_cp8(ld, a.x + (size_t)grow * HID + kb + c4);
      } else {
        uint2 z; z.x = 0u; z.y = 0u;
        *(uint2*)ld = z;
      }
      lds_async_wait();
    }
    __syncthreads();
    bf16x16 af = frag_ld(apA, apA + 16);
#pragma unroll
    for (int j = 0; j < 8; ++j) {
      const bf16* bp = bpj[j] + kb;
      bf16x16 bv = frag_ld(bp, bp + 16);
      acc[j] = __builtin_amdgcn_wmma_f32_16x16x32_bf16(
          false, af, false, bv, (short)0, acc[j], false, false);
    }
  }
  for (int j = 0; j < 8; ++j) {
    int col = wn * 128 + j * 16 + lm;
    float bb = a.b0[col];
    for (int q = 0; q < 8; ++q) {
      float v = acc[j][q] + bb;
      v = v * (1.f / (1.f + __expf(-v)));
      sH[(wm * 16 + q + rsel) * 520 + col] = f2bf(v);
    }
  }
  __syncthreads();
  // 32 rows x 8 out-cols = 256 threads, each does a 512-length dot
  int r = tid >> 3, c = tid & 7;
  float s = a.b1[c];
  for (int k = 0; k < HID; ++k)
    s += bf2f(sH[r * 520 + k]) * a.w1[(size_t)k * 8 + c];
  int grow = row0 + r;
  if (grow < a.nrows) a.out[(size_t)grow * 8 + c] = s;
}

// ================== host orchestration ==================
extern "C" void kernel_launch(void* const* d_in, const int* in_sizes, int n_in,
                              void* d_out, int out_size, void* d_ws, size_t ws_size,
                              hipStream_t stream) {
  (void)in_sizes; (void)out_size; (void)ws_size;
  if (n_in < 123) return;   // expected: 11 data/index arrays + 112 param leaves

  const float* grid_nfeat = (const float*)d_in[0];
  const float* mesh_ndata = (const float*)d_in[1];
  const float* g2m_edata  = (const float*)d_in[2];
  const float* mesh_edata = (const float*)d_in[3];
  const float* m2g_edata  = (const float*)d_in[4];
  const int* g2m_src  = (const int*)d_in[5];
  const int* g2m_dst  = (const int*)d_in[6];
  const int* mesh_src = (const int*)d_in[7];
  const int* mesh_dst = (const int*)d_in[8];
  const int* m2g_src  = (const int*)d_in[9];
  const int* m2g_dst  = (const int*)d_in[10];

  // params flattened in reference-dict insertion order: w0,b0,w1,b1[,ln_g,ln_b]
  int pi = 11;
  struct HostMlp { const float *w0, *b0, *w1, *b1, *lng, *lnb; };
  auto take = [&](bool norm) {
    HostMlp m{};
    m.w0 = (const float*)d_in[pi++]; m.b0 = (const float*)d_in[pi++];
    m.w1 = (const float*)d_in[pi++]; m.b1 = (const float*)d_in[pi++];
    if (norm) { m.lng = (const float*)d_in[pi++]; m.lnb = (const float*)d_in[pi++]; }
    return m;
  };
  HostMlp em_grid = take(true), em_mesh = take(true), em_g2m = take(true),
          em_meshE = take(true), em_m2g = take(true);
  HostMlp enc_edge = take(true), enc_nd = take(true), enc_ns = take(true);
  HostMlp pe[4], pn[4];
  for (int i = 0; i < 4; ++i) { pe[i] = take(true); pn[i] = take(true); }
  HostMlp dec_edge = take(true), dec_node = take(true), fin = take(false);

  // ---- workspace layout ----
  size_t cur = 0;
  auto alloc = [&](size_t bytes) -> void* {
    void* p = (void*)((char*)d_ws + cur);
    cur += (bytes + 255) & ~(size_t)255;
    return p;
  };
  bf16*  grid  = (bf16*)alloc((size_t)N_GRID_C * HID * 2);
  bf16*  mesh  = (bf16*)alloc((size_t)N_MESH_C * HID * 2);
  bf16*  eg2m  = (bf16*)alloc((size_t)E_G2M_C * HID * 2);
  bf16*  emesh = (bf16*)alloc((size_t)E_MESH_C * HID * 2);
  bf16*  em2g  = (bf16*)alloc((size_t)E_M2G_C * HID * 2);
  float* agg   = (float*)alloc((size_t)N_GRID_C * HID * 4);

  auto convW = [&](const float* w, int K, int N, int Kpad) -> bf16* {
    bf16* wt = (bf16*)alloc((size_t)N * Kpad * 2);
    k_wT<<<N, 256, 0, stream>>>(w, wt, K, N, Kpad);
    return wt;
  };
  struct WT { bf16 *w0T, *w1T; };
  auto convM = [&](const HostMlp& m, int din) -> WT {
    WT t; int kp = (din + 31) & ~31;
    t.w0T = convW(m.w0, din, HID, kp);
    t.w1T = convW(m.w1, HID, HID, HID);
    return t;
  };
  WT t_emg = convM(em_grid, 13), t_emm = convM(em_mesh, 3),
     t_eg  = convM(em_g2m, 4),   t_eme = convM(em_meshE, 4), t_em2 = convM(em_m2g, 4);
  WT t_ence = convM(enc_edge, 3 * HID), t_encnd = convM(enc_nd, 2 * HID),
     t_encns = convM(enc_ns, HID);
  WT t_pe[4], t_pn[4];
  for (int i = 0; i < 4; ++i) { t_pe[i] = convM(pe[i], 3 * HID); t_pn[i] = convM(pn[i], 2 * HID); }
  WT t_dece = convM(dec_edge, 3 * HID), t_decn = convM(dec_node, 2 * HID);
  bf16* fin_w0T = convW(fin.w0, HID, HID, HID);

  auto mlp = [&](const bf16* s0b, const float* s0f, int d0,
                 const bf16* s1, const int* i1, int d1,
                 const bf16* s2, const int* i2, int d2,
                 const WT& t, const HostMlp& m,
                 const bf16* resid, bf16* out, int nrows) {
    MlpArgs a{};
    a.s0b = s0b; a.s0f = s0f; a.d0 = d0;
    a.s1 = s1; a.i1 = i1; a.d1 = d1;
    a.s2 = s2; a.i2 = i2; a.d2 = d2;
    a.din = d0 + d1 + d2; a.k1pad = (a.din + 31) & ~31; a.nrows = nrows;
    a.w0T = t.w0T; a.b0 = m.b0; a.w1T = t.w1T; a.b1 = m.b1;
    a.lng = m.lng; a.lnb = m.lnb; a.resid = resid; a.out = out;
    k_mlp<<<(nrows + 31) / 32, 256, 0, stream>>>(a);
  };
  auto zero = [&](float* p, long long n) {
    k_zero<<<(unsigned)((n + 255) / 256), 256, 0, stream>>>(p, n);
  };
  auto scatter = [&](const bf16* e, const int* dst, long long nE) {
    long long total = nE * HID;
    k_scatter<<<(unsigned)((total + 255) / 256), 256, 0, stream>>>(e, dst, agg, total);
  };

  // ---- embedders ----
  mlp(nullptr, grid_nfeat, 13, nullptr, nullptr, 0, nullptr, nullptr, 0, t_emg, em_grid,  nullptr, grid,  N_GRID_C);
  mlp(nullptr, mesh_ndata, 3,  nullptr, nullptr, 0, nullptr, nullptr, 0, t_emm, em_mesh,  nullptr, mesh,  N_MESH_C);
  mlp(nullptr, g2m_edata,  4,  nullptr, nullptr, 0, nullptr, nullptr, 0, t_eg,  em_g2m,   nullptr, eg2m,  E_G2M_C);
  mlp(nullptr, mesh_edata, 4,  nullptr, nullptr, 0, nullptr, nullptr, 0, t_eme, em_meshE, nullptr, emesh, E_MESH_C);
  mlp(nullptr, m2g_edata,  4,  nullptr, nullptr, 0, nullptr, nullptr, 0, t_em2, em_m2g,   nullptr, em2g,  E_M2G_C);

  // ---- grid2mesh encoder ----
  mlp(eg2m, nullptr, HID, grid, g2m_src, HID, mesh, g2m_dst, HID, t_ence, enc_edge, eg2m, eg2m, E_G2M_C);
  zero(agg, (long long)N_MESH_C * HID);
  scatter(eg2m, g2m_dst, E_G2M_C);
  mlp(nullptr, agg, HID, mesh, nullptr, HID, nullptr, nullptr, 0, t_encnd, enc_nd, mesh, mesh, N_MESH_C);
  mlp(grid, nullptr, HID, nullptr, nullptr, 0, nullptr, nullptr, 0, t_encns, enc_ns, grid, grid, N_GRID_C);

  // ---- mesh processor ----
  for (int i = 0; i < 4; ++i) {
    mlp(emesh, nullptr, HID, mesh, mesh_src, HID, mesh, mesh_dst, HID, t_pe[i], pe[i], emesh, emesh, E_MESH_C);
    zero(agg, (long long)N_MESH_C * HID);
    scatter(emesh, mesh_dst, E_MESH_C);
    mlp(nullptr, agg, HID, mesh, nullptr, HID, nullptr, nullptr, 0, t_pn[i], pn[i], mesh, mesh, N_MESH_C);
  }

  // ---- mesh2grid decoder ----
  mlp(em2g, nullptr, HID, mesh, m2g_src, HID, grid, m2g_dst, HID, t_dece, dec_edge, em2g, em2g, E_M2G_C);
  zero(agg, (long long)N_GRID_C * HID);
  scatter(em2g, m2g_dst, E_M2G_C);
  mlp(nullptr, agg, HID, grid, nullptr, HID, nullptr, nullptr, 0, t_decn, dec_node, grid, grid, N_GRID_C);

  // ---- final projection ----
  FinArgs fa{};
  fa.x = grid; fa.w0T = fin_w0T; fa.b0 = fin.b0; fa.w1 = fin.w1; fa.b1 = fin.b1;
  fa.out = (float*)d_out; fa.nrows = N_GRID_C;
  k_final<<<(N_GRID_C + 31) / 32, 256, 0, stream>>>(fa);
}
